// GanaGAT_27522150433353
// MI455X (gfx1250) — compile-verified, matrix operand order checked by reading, not run on hardware
//
#include <hip/hip_runtime.h>
#include <hip/hip_bf16.h>

#define NNODES 10000
#define NEDGES 160000
#define EN_EDGES (NEDGES + NNODES)
#define NEG_SLOPE 0.2f

typedef __attribute__((ext_vector_type(16))) __bf16    v16bf;
typedef __attribute__((ext_vector_type(8)))  float     v8f;
typedef __attribute__((ext_vector_type(4)))  unsigned  v4u;
typedef __attribute__((ext_vector_type(8)))  int       v8i;
typedef __attribute__((ext_vector_type(4)))  int       v4i;

// ---------- helpers ----------
__device__ __forceinline__ unsigned short f2bf(float f) {
    unsigned u = __float_as_uint(f);
    u += 0x7FFFu + ((u >> 16) & 1u);     // round-to-nearest-even
    return (unsigned short)(u >> 16);
}
// monotone float->uint key for atomicMax-based segment max
__device__ __forceinline__ unsigned fkey(float f) {
    unsigned u = __float_as_uint(f);
    return (u & 0x80000000u) ? ~u : (u | 0x80000000u);
}
__device__ __forceinline__ float fdec(unsigned k) {
    unsigned u = (k & 0x80000000u) ? (k & 0x7FFFFFFFu) : ~k;
    return __uint_as_float(u);
}
__device__ __forceinline__ void edge_nodes(const int* ei, int e, int& s, int& d) {
    if (e < NEDGES) { s = ei[e]; d = ei[NEDGES + e]; }
    else            { s = d = e - NEDGES; }              // appended self-loops
}

// ---------- TDM: 2D tile (rows x 32 bf16) global -> LDS, per ISA D# layout ----------
// group0: [1:0]=count(1), [63:32]=lds_addr, [120:64]=global_addr, [127:126]=type(2)
// group1: [17:16]=data_size(1 => 2B), [79:48]=tensor_dim0, [111:80]=tensor_dim1,
//         [127:112]=tile_dim0, [143:128]=tile_dim1, [207:160]=tensor_dim0_stride
__device__ __forceinline__ void tdm_load_tile(unsigned lds_off, const void* gaddr,
                                              unsigned tile_rows, unsigned stride_elems) {
    unsigned long long ga = (unsigned long long)gaddr;
    const unsigned td0 = 32u;            // contiguous K elements per row
    v4u g0;
    g0[0] = 1u;                                                  // count=1, user desc
    g0[1] = lds_off;                                             // lds_addr (bytes)
    g0[2] = (unsigned)(ga & 0xFFFFFFFFu);                        // global_addr lo
    g0[3] = (unsigned)((ga >> 32) & 0x01FFFFFFu) | (2u << 30);   // addr[56:32] | type=2
    v8i g1;
    g1[0] = (int)(1u << 16);                                     // data_size = 2 bytes
    g1[1] = (int)((td0 & 0xFFFFu) << 16);                        // tensor_dim0 lo16
    g1[2] = (int)(((td0 >> 16) & 0xFFFFu) | ((tile_rows & 0xFFFFu) << 16)); // td0 hi | td1 lo
    g1[3] = (int)(((tile_rows >> 16) & 0xFFFFu) | (32u << 16));  // td1 hi | tile_dim0=32
    g1[4] = (int)(tile_rows & 0xFFFFu);                          // tile_dim1 (tile_dim2=0)
    g1[5] = (int)stride_elems;                                   // tensor_dim0_stride lo32
    g1[6] = 0;
    g1[7] = 0;
    v4i gz; gz[0] = 0; gz[1] = 0; gz[2] = 0; gz[3] = 0;          // groups 2/3 unused (2D)
#if __clang_major__ >= 23
    v8i gz8; gz8[0]=0; gz8[1]=0; gz8[2]=0; gz8[3]=0; gz8[4]=0; gz8[5]=0; gz8[6]=0; gz8[7]=0;
    __builtin_amdgcn_tensor_load_to_lds(g0, g1, gz, gz, gz8, 0);
#else
    __builtin_amdgcn_tensor_load_to_lds(g0, g1, gz, gz, 0);
#endif
}

// ---------- weight transpose + bf16 convert: W[K,N] -> Wt[Npad,Kpad] ----------
__global__ void wt_convert_kernel(const float* __restrict__ W, unsigned short* __restrict__ Wt,
                                  int K, int Nv, int Kpad, int Npad) {
    long idx = (long)blockIdx.x * blockDim.x + threadIdx.x;
    long total = (long)Npad * Kpad;
    if (idx >= total) return;
    int kp = (int)(idx % Kpad);
    int np = (int)(idx / Kpad);
    float v = (kp < K && np < Nv) ? W[(size_t)kp * Nv + np] : 0.0f;
    Wt[idx] = f2bf(v);
}

// ---------- x[N,50] -> bf16 [N,64] zero padded ----------
__global__ void x_convert_kernel(const float* __restrict__ x, unsigned short* __restrict__ xb,
                                 int F, int Kpad) {
    long idx = (long)blockIdx.x * blockDim.x + threadIdx.x;
    long total = (long)NNODES * Kpad;
    if (idx >= total) return;
    int kp = (int)(idx % Kpad);
    int n  = (int)(idx / Kpad);
    float v = (kp < F) ? x[(size_t)n * F + kp] : 0.0f;
    xb[idx] = f2bf(v);
}

// ---------- bf16 WMMA GEMM with TDM double-buffered LDS staging ----------
// C[M,Npad] = A[M,Kpad] * Bt[Npad,Kpad]^T
// Block = 256 threads (8 waves) -> 128(M) x 64(N) tile; wave = 16(M) x 64(N).
// Wave 0 drives tensor_load_to_lds for the shared A(128x32) / B(64x32) K-step tiles.
__global__ void __launch_bounds__(256)
gemm_bf16_wmma_kernel(const unsigned short* __restrict__ A,
                      const unsigned short* __restrict__ Bt,
                      float* __restrict__ C,
                      int M, int Npad, int Kpad) {
    __shared__ __align__(16) unsigned short sA[2][128 * 32];   // 2 x 8KB
    __shared__ __align__(16) unsigned short sB[2][64 * 32];    // 2 x 4KB

    const int lane = threadIdx.x & 31;
    const int wave = threadIdx.x >> 5;
    const int half = lane >> 4;        // K-half selector per 16-bit fragment layout
    const int r    = lane & 15;
    const int blockM = blockIdx.x * 128;
    const int tileM  = blockM + wave * 16;
    const int tileN  = blockIdx.y * 64;
    const bool active = (tileM < M);   // wave-uniform; wave 0 always active
    const bool tdmWave = (wave == 0);

    const unsigned ldsA0 = (unsigned)(unsigned long long)(void*)&sA[0][0];
    const unsigned ldsA1 = (unsigned)(unsigned long long)(void*)&sA[1][0];
    const unsigned ldsB0 = (unsigned)(unsigned long long)(void*)&sB[0][0];
    const unsigned ldsB1 = (unsigned)(unsigned long long)(void*)&sB[1][0];

    const int steps = Kpad >> 5;       // K-steps of 32 (Kpad is 64 or 1024)
    const unsigned short* Ab = A + (size_t)blockM * Kpad;
    const unsigned short* Bb = Bt + (size_t)tileN * Kpad;

    // prologue: prefill both LDS buffers (pairs for steps 0 and 1)
    if (tdmWave) {
        tdm_load_tile(ldsA0, Ab,      128, (unsigned)Kpad);
        tdm_load_tile(ldsB0, Bb,       64, (unsigned)Kpad);
        tdm_load_tile(ldsA1, Ab + 32, 128, (unsigned)Kpad);
        tdm_load_tile(ldsB1, Bb + 32,  64, (unsigned)Kpad);
    }

    v8f acc[4];
    acc[0] = v8f{}; acc[1] = v8f{}; acc[2] = v8f{}; acc[3] = v8f{};

    for (int k = 0; k < steps; ++k) {
        const int buf = k & 1;
        if (tdmWave) {
            if (k + 1 < steps) __builtin_amdgcn_s_wait_tensorcnt(2); // pair k complete
            else               __builtin_amdgcn_s_wait_tensorcnt(0);
        }
        __syncthreads();

        if (active) {
            // Stage all fragments first so the 10 ds_load_b128 pipeline ahead of the MMAs.
            // A fragment: lane holds row (wave*16+r); elems 0..7 = K[8h..8h+7],
            // elems 8..15 = K[16+8h .. 16+8h+7]  (ISA 16-bit A 16x32 layout)
            const unsigned short* aRow  = &sA[buf][(wave * 16 + r) * 32];
            const unsigned short* bBase = &sB[buf][r * 32];
            v16bf afrag, bfrag[4];
            __builtin_memcpy(&afrag,             aRow + half * 8,      16);
            __builtin_memcpy((char*)&afrag + 16, aRow + 16 + half * 8, 16);
#pragma unroll
            for (int nt = 0; nt < 4; ++nt) {
                const unsigned short* bRow = bBase + nt * (16 * 32);
                __builtin_memcpy(&bfrag[nt],             bRow + half * 8,      16);
                __builtin_memcpy((char*)&bfrag[nt] + 16, bRow + 16 + half * 8, 16);
            }
#pragma unroll
            for (int nt = 0; nt < 4; ++nt) {
                acc[nt] = __builtin_amdgcn_wmma_f32_16x16x32_bf16(
                    false, afrag, false, bfrag[nt], (short)0, acc[nt], false, false);
            }
        }
        __syncthreads();   // everyone done reading buf before TDM overwrites it

        if (tdmWave && (k + 2 < steps)) {
            const int kk = (k + 2) * 32;
            tdm_load_tile(buf ? ldsA1 : ldsA0, Ab + kk, 128, (unsigned)Kpad);
            tdm_load_tile(buf ? ldsB1 : ldsB0, Bb + kk,  64, (unsigned)Kpad);
        }
    }

    if (active) {
        // C/D layout: VGPR v -> M = v + 8*half, N = tileN + nt*16 + r
#pragma unroll
        for (int v = 0; v < 8; ++v) {
            const int row = tileM + v + 8 * half;
            float* Cr = C + (size_t)row * Npad + tileN + r;
            Cr[0]  = acc[0][v];
            Cr[16] = acc[1][v];
            Cr[32] = acc[2][v];
            Cr[48] = acc[3][v];
        }
    }
}

// ---------- attention scores: s_src[n,h] = sum_c h[n,h,c]*a_src[h,c] ----------
__global__ void scores_kernel(const float* __restrict__ h,
                              const float* __restrict__ a_src, const float* __restrict__ a_dst,
                              float* __restrict__ s_src, float* __restrict__ s_dst,
                              int H, int C, int ld) {
    __shared__ float ss[8], sd[8];
    const int n = blockIdx.x;
    const int tid = threadIdx.x;
    if (tid < H) { ss[tid] = 0.0f; sd[tid] = 0.0f; }
    __syncthreads();
    const float* row = h + (size_t)n * ld;
    const int HC = H * C;
    for (int idx = tid; idx < HC; idx += blockDim.x) {
        float v = row[idx];
        int hh = idx / C;
        atomicAdd(&ss[hh], v * a_src[idx]);
        atomicAdd(&sd[hh], v * a_dst[idx]);
    }
    __syncthreads();
    if (tid < H) { s_src[(size_t)n * H + tid] = ss[tid]; s_dst[(size_t)n * H + tid] = sd[tid]; }
}

// ---------- edge pass 1: alpha = leaky_relu(...), segment max via key atomicMax ----------
__global__ void edge_max_kernel(const int* __restrict__ ei,
                                const float* __restrict__ s_src, const float* __restrict__ s_dst,
                                float* __restrict__ alpha, unsigned* __restrict__ m_u, int H) {
    long idx = (long)blockIdx.x * blockDim.x + threadIdx.x;
    long total = (long)EN_EDGES * H;
    if (idx >= total) return;
    int hh = (int)(idx % H);
    int e  = (int)(idx / H);
    int s, d; edge_nodes(ei, e, s, d);
    float a = s_src[(size_t)s * H + hh] + s_dst[(size_t)d * H + hh];
    a = (a > 0.0f) ? a : NEG_SLOPE * a;
    alpha[idx] = a;
    atomicMax(&m_u[(size_t)d * H + hh], fkey(a));
}

// ---------- edge pass 2: e = exp(alpha - max); denom += e ----------
__global__ void edge_exp_kernel(const int* __restrict__ ei,
                                float* __restrict__ alpha, const unsigned* __restrict__ m_u,
                                float* __restrict__ denom, int H) {
    long idx = (long)blockIdx.x * blockDim.x + threadIdx.x;
    long total = (long)EN_EDGES * H;
    if (idx >= total) return;
    int hh = (int)(idx % H);
    int e  = (int)(idx / H);
    int s, d; edge_nodes(ei, e, s, d);
    float m = fdec(m_u[(size_t)d * H + hh]);
    float ev = __expf(alpha[idx] - m);
    alpha[idx] = ev;
    atomicAdd(&denom[(size_t)d * H + hh], ev);
}

// ---------- edge pass 3: coef = e / denom[dst] ----------
__global__ void edge_coef_kernel(const int* __restrict__ ei,
                                 float* __restrict__ alpha, const float* __restrict__ denom, int H) {
    long idx = (long)blockIdx.x * blockDim.x + threadIdx.x;
    long total = (long)EN_EDGES * H;
    if (idx >= total) return;
    int hh = (int)(idx % H);
    int e  = (int)(idx / H);
    int s, d; edge_nodes(ei, e, s, d);
    alpha[idx] = alpha[idx] / denom[(size_t)d * H + hh];
}

// ---------- edge pass 4: out[dst] += coef * h[src]  (L2-resident f32 atomics) ----------
__global__ void edge_aggr_kernel(const int* __restrict__ ei,
                                 const float* __restrict__ coef, const float* __restrict__ h,
                                 float* __restrict__ out, int H, int C, int CT) {
    long idx = (long)blockIdx.x * blockDim.x + threadIdx.x;
    long total = (long)EN_EDGES * CT;
    if (idx >= total) return;
    int c = (int)(idx % CT);
    int e = (int)(idx / CT);
    int s, d; edge_nodes(ei, e, s, d);
    int hh = c / C;
    float w = coef[(size_t)e * H + hh];
    atomicAdd(&out[(size_t)d * CT + c], w * h[(size_t)s * CT + c]);
}

// ---------- bias + ELU + bf16 for next layer's A ----------
__global__ void epilogue_elu_kernel(const float* __restrict__ obuf, const float* __restrict__ bias,
                                    unsigned short* __restrict__ abuf, int D) {
    long idx = (long)blockIdx.x * blockDim.x + threadIdx.x;
    long total = (long)NNODES * D;
    if (idx >= total) return;
    int dd = (int)(idx % D);
    float v = obuf[idx] + bias[dd];
    v = (v > 0.0f) ? v : (__expf(v) - 1.0f);
    abuf[idx] = f2bf(v);
}

// ---------- final: d_out[n,c] = obuf[n*128+c] + b3[c] (pre-ELU) ----------
__global__ void final_kernel(const float* __restrict__ obuf, const float* __restrict__ b3,
                             float* __restrict__ out, int NC, int ld) {
    long idx = (long)blockIdx.x * blockDim.x + threadIdx.x;
    long total = (long)NNODES * NC;
    if (idx >= total) return;
    int c = (int)(idx % NC);
    int n = (int)(idx / NC);
    out[idx] = obuf[(size_t)n * ld + c] + b3[c];
}

// =====================================================================
extern "C" void kernel_launch(void* const* d_in, const int* in_sizes, int n_in,
                              void* d_out, int out_size, void* d_ws, size_t ws_size,
                              hipStream_t stream) {
    const float* x      = (const float*)d_in[0];
    const int*   ei     = (const int*)  d_in[1];
    const float* W1     = (const float*)d_in[2];
    const float* a_src1 = (const float*)d_in[3];
    const float* a_dst1 = (const float*)d_in[4];
    const float* b1     = (const float*)d_in[5];
    const float* W2     = (const float*)d_in[6];
    const float* a_src2 = (const float*)d_in[7];
    const float* a_dst2 = (const float*)d_in[8];
    const float* b2     = (const float*)d_in[9];
    const float* W3     = (const float*)d_in[10];
    const float* a_src3 = (const float*)d_in[11];
    const float* a_dst3 = (const float*)d_in[12];
    const float* b3     = (const float*)d_in[13];
    float* out = (float*)d_out;

    // ---- workspace carve-out (256B aligned) ----
    char* ws = (char*)d_ws;
    size_t off = 0;
    auto alloc = [&](size_t bytes) -> char* {
        char* p = ws + off;
        off = (off + bytes + 255) & ~(size_t)255;
        return p;
    };
    float*          hbuf  = (float*)         alloc((size_t)NNODES * 1024 * 4); // GEMM out
    float*          obuf  = (float*)         alloc((size_t)NNODES * 1024 * 4); // aggregation
    unsigned short* abuf  = (unsigned short*)alloc((size_t)NNODES * 1024 * 2); // bf16 A (next layer)
    unsigned short* xb    = (unsigned short*)alloc((size_t)NNODES * 64 * 2);
    unsigned short* W1t   = (unsigned short*)alloc((size_t)1024 * 64 * 2);
    unsigned short* W2t   = (unsigned short*)alloc((size_t)1024 * 1024 * 2);
    unsigned short* W3t   = (unsigned short*)alloc((size_t)128 * 1024 * 2);
    float*          ssrc  = (float*)         alloc((size_t)NNODES * 8 * 4);
    float*          sdst  = (float*)         alloc((size_t)NNODES * 8 * 4);
    unsigned*       m_u   = (unsigned*)      alloc((size_t)NNODES * 8 * 4);
    float*          denom = (float*)         alloc((size_t)NNODES * 8 * 4);
    float*          alpha = (float*)         alloc((size_t)EN_EDGES * 8 * 4);

    const int T = 256;
    auto nblk = [](long total, int t) { return (unsigned)((total + t - 1) / t); };

    // ---- one-time converts ----
    wt_convert_kernel<<<nblk((long)1024 * 64, T), T, 0, stream>>>(W1, W1t, 50, 1024, 64, 1024);
    wt_convert_kernel<<<nblk((long)1024 * 1024, T), T, 0, stream>>>(W2, W2t, 1024, 1024, 1024, 1024);
    wt_convert_kernel<<<nblk((long)128 * 1024, T), T, 0, stream>>>(W3, W3t, 1024, 121, 1024, 128);
    x_convert_kernel<<<nblk((long)NNODES * 64, T), T, 0, stream>>>(x, xb, 50, 64);

    const unsigned gemmBlocksM = (NNODES + 127) / 128; // 79

    // ================= Layer 1: 50 -> 8x128, concat =================
    {
        const int H = 8, C = 128, CT = 1024;
        gemm_bf16_wmma_kernel<<<dim3(gemmBlocksM, 1024 / 64), T, 0, stream>>>(
            xb, W1t, hbuf, NNODES, 1024, 64);
        scores_kernel<<<NNODES, T, 0, stream>>>(hbuf, a_src1, a_dst1, ssrc, sdst, H, C, CT);
        hipMemsetAsync(m_u,   0, (size_t)NNODES * H * 4, stream);
        hipMemsetAsync(denom, 0, (size_t)NNODES * H * 4, stream);
        hipMemsetAsync(obuf,  0, (size_t)NNODES * CT * 4, stream);
        long eh = (long)EN_EDGES * H;
        edge_max_kernel <<<nblk(eh, T), T, 0, stream>>>(ei, ssrc, sdst, alpha, m_u, H);
        edge_exp_kernel <<<nblk(eh, T), T, 0, stream>>>(ei, alpha, m_u, denom, H);
        edge_coef_kernel<<<nblk(eh, T), T, 0, stream>>>(ei, alpha, denom, H);
        edge_aggr_kernel<<<nblk((long)EN_EDGES * CT, T), T, 0, stream>>>(ei, alpha, hbuf, obuf, H, C, CT);
        epilogue_elu_kernel<<<nblk((long)NNODES * CT, T), T, 0, stream>>>(obuf, b1, abuf, CT);
    }

    // ================= Layer 2: 1024 -> 1024, 1 head =================
    {
        const int H = 1, C = 1024, CT = 1024;
        gemm_bf16_wmma_kernel<<<dim3(gemmBlocksM, 1024 / 64), T, 0, stream>>>(
            abuf, W2t, hbuf, NNODES, 1024, 1024);
        scores_kernel<<<NNODES, T, 0, stream>>>(hbuf, a_src2, a_dst2, ssrc, sdst, H, C, CT);
        hipMemsetAsync(m_u,   0, (size_t)NNODES * H * 4, stream);
        hipMemsetAsync(denom, 0, (size_t)NNODES * H * 4, stream);
        hipMemsetAsync(obuf,  0, (size_t)NNODES * CT * 4, stream);
        long eh = (long)EN_EDGES * H;
        edge_max_kernel <<<nblk(eh, T), T, 0, stream>>>(ei, ssrc, sdst, alpha, m_u, H);
        edge_exp_kernel <<<nblk(eh, T), T, 0, stream>>>(ei, alpha, m_u, denom, H);
        edge_coef_kernel<<<nblk(eh, T), T, 0, stream>>>(ei, alpha, denom, H);
        edge_aggr_kernel<<<nblk((long)EN_EDGES * CT, T), T, 0, stream>>>(ei, alpha, hbuf, obuf, H, C, CT);
        epilogue_elu_kernel<<<nblk((long)NNODES * CT, T), T, 0, stream>>>(obuf, b2, abuf, CT);
    }

    // ============ Layer 3: 1024 -> 121 (N padded to 128), mean(1 head) ============
    {
        const int H = 1, C = 128, CT = 128; // pad cols 121..127 are exact zeros (zero W rows)
        gemm_bf16_wmma_kernel<<<dim3(gemmBlocksM, 128 / 64), T, 0, stream>>>(
            abuf, W3t, hbuf, NNODES, 128, 1024);
        scores_kernel<<<NNODES, T, 0, stream>>>(hbuf, a_src3, a_dst3, ssrc, sdst, H, 121, 128);
        hipMemsetAsync(m_u,   0, (size_t)NNODES * H * 4, stream);
        hipMemsetAsync(denom, 0, (size_t)NNODES * H * 4, stream);
        hipMemsetAsync(obuf,  0, (size_t)NNODES * CT * 4, stream);
        long eh = (long)EN_EDGES * H;
        edge_max_kernel <<<nblk(eh, T), T, 0, stream>>>(ei, ssrc, sdst, alpha, m_u, H);
        edge_exp_kernel <<<nblk(eh, T), T, 0, stream>>>(ei, alpha, m_u, denom, H);
        edge_coef_kernel<<<nblk(eh, T), T, 0, stream>>>(ei, alpha, denom, H);
        edge_aggr_kernel<<<nblk((long)EN_EDGES * CT, T), T, 0, stream>>>(ei, alpha, hbuf, obuf, H, C, CT);
        final_kernel<<<nblk((long)NNODES * 121, T), T, 0, stream>>>(obuf, b3, out, 121, 128);
    }
}
